// PatchedAttentionBlock_24824910970984
// MI455X (gfx1250) — matrix operand order repaired and checked by reading, not compiled, not run on hardware
//
#include <hip/hip_runtime.h>

// ---------------------------------------------------------------------------
// Problem constants: NF=64, PE=4, NPX=8, K=8, B=4, S=32, D=4096, DK=1024
// N = 2048 query patches; kv rows = 16384.
// ---------------------------------------------------------------------------

typedef __attribute__((ext_vector_type(16))) __bf16 v16bf;
typedef __attribute__((ext_vector_type(8)))  __bf16 v8bf;
typedef __attribute__((ext_vector_type(8)))  float  v8f;
typedef __attribute__((ext_vector_type(4)))  unsigned int v4u;
typedef __attribute__((ext_vector_type(8)))  int v8i;
typedef __attribute__((ext_vector_type(4)))  int v4i;

#if defined(__has_builtin)
#if __has_builtin(__builtin_amdgcn_tensor_load_to_lds) && \
    __has_builtin(__builtin_amdgcn_s_wait_tensorcnt)
#define USE_TDM 1
#endif
#endif
#ifndef USE_TDM
#define USE_TDM 0
#endif

__device__ __forceinline__ unsigned short f32_to_bf16(float f) {
  union { float f; unsigned u; } v; v.f = f;
  unsigned r = v.u + 0x7fffu + ((v.u >> 16) & 1u);   // round-to-nearest-even
  return (unsigned short)(r >> 16);
}
__device__ __forceinline__ float bf16_to_f32(unsigned short s) {
  union { unsigned u; float f; } v; v.u = ((unsigned)s) << 16;
  return v.f;
}

// ---------------------------------------------------------------------------
// Tiled transpose + fp32->bf16: src (Kd x N) row-major -> dst (N x Kd) bf16.
// 32x32 tiles, coalesced loads and stores, LDS padded 33 to avoid conflicts.
// ---------------------------------------------------------------------------
__global__ __launch_bounds__(256)
void transpose_cvt_kernel(const float* __restrict__ src,
                          unsigned short* __restrict__ dst,
                          int Kd, int N) {
  __shared__ unsigned short tile[32][33];
  const int n0 = blockIdx.x * 32, k0 = blockIdx.y * 32;
  const int tx = threadIdx.x, ty = threadIdx.y;   // 32 x 8
#pragma unroll
  for (int r = 0; r < 4; ++r) {
    int k = k0 + ty + 8 * r;
    tile[ty + 8 * r][tx] = f32_to_bf16(src[(size_t)k * N + n0 + tx]);
  }
  __syncthreads();
#pragma unroll
  for (int r = 0; r < 4; ++r) {
    int n = n0 + ty + 8 * r;
    dst[(size_t)n * Kd + k0 + tx] = tile[tx][ty + 8 * r];
  }
}

// ---------------------------------------------------------------------------
// Unfold x_predicted -> Apred (2048 x 4096) bf16
// ---------------------------------------------------------------------------
__global__ void pack_pred_kernel(const float* __restrict__ xp,
                                 unsigned short* __restrict__ Apred) {
  int idx = blockIdx.x * 256 + threadIdx.x;          // [0, 2048*4096)
  int d = idx & 4095;
  int n = idx >> 12;
  int p3 = n & 7, p2 = (n >> 3) & 7, p1 = (n >> 6) & 7, b = n >> 9;
  int e3 = d & 3, e2 = (d >> 2) & 3, e1 = (d >> 4) & 3, nf = d >> 6;
  int s1 = p1 * 4 + e1, s2 = p2 * 4 + e2, s3 = p3 * 4 + e3;
  int src = ((b * 64 + nf) * 32 + s1) * 1024 + s2 * 32 + s3;
  Apred[idx] = f32_to_bf16(xp[src]);
}

// ---------------------------------------------------------------------------
// Unfold x_retrieved -> Akv (16384 x 4096) bf16, row = n*8 + k
// ---------------------------------------------------------------------------
__global__ void pack_kv_kernel(const float* __restrict__ xr,
                               unsigned short* __restrict__ Akv) {
  int idx = blockIdx.x * 256 + threadIdx.x;          // [0, 16384*4096)
  int d = idx & 4095;
  int row = idx >> 12;
  int k = row & 7;
  int n = row >> 3;
  int p3 = n & 7, p2 = (n >> 3) & 7, p1 = (n >> 6) & 7, b = n >> 9;
  int e3 = d & 3, e2 = (d >> 2) & 3, e1 = (d >> 4) & 3, nf = d >> 6;
  int s1 = p1 * 4 + e1, s2 = p2 * 4 + e2, s3 = p3 * 4 + e3;
  int src = (((b * 8 + k) * 64 + nf) * 32 + s1) * 1024 + s2 * 32 + s3;
  Akv[idx] = f32_to_bf16(xr[src]);
}

#if USE_TDM
// ---------------------------------------------------------------------------
// Issue one TDM 2D tile load: 32 (contiguous K, bf16) x 128 rows, row stride
// Kd elements, into LDS with 16B padding after every 64B row (matches the
// [128][40]-ushort LDS layout). D# packing per CDNA5 ISA 8.3/8.4.
// This toolchain exposes the 6-arg builtin:
//   (uint32x4 g0, int32x8 g1, int32x4 g2, int32x4 g3, int32x8, i32 cpol)
// ---------------------------------------------------------------------------
__device__ __forceinline__ void tdm_issue_tile(unsigned long long lds_flat,
                                               const unsigned short* gptr,
                                               int Kd) {
  unsigned long long ga = (unsigned long long)gptr;
  v4u g0;
  g0[0] = 1u;                                   // count=1, user load
  g0[1] = (unsigned)lds_flat;                   // lds_addr (low 32 = LDS offset)
  g0[2] = (unsigned)ga;                         // global_addr[31:0]
  g0[3] = (unsigned)(ga >> 32) | 0x80000000u;   // global_addr[56:32] | type=2
  v8i g1;
  // wg_mask=0 | data_size=1 (2B) | pad_enable | pad_interval=3 (16 dw=64B)
  // | pad_amount=3 (4 dw = 16B)
  g1[0] = (1 << 16) | (1 << 20) | (3 << 22) | (3 << 25);
  const unsigned td0 = (unsigned)Kd;            // tensor_dim0
  const unsigned td1 = 65536u;                  // tensor_dim1 (ample)
  g1[1] = (int)((td0 & 0xffffu) << 16);                   // dim0[15:0]
  g1[2] = (int)((td0 >> 16) | ((td1 & 0xffffu) << 16));   // dim0[31:16]|dim1[15:0]
  g1[3] = (int)((td1 >> 16) | (32u << 16));               // dim1[31:16]|tile_dim0=32
  g1[4] = 128;                                  // tile_dim1=128 (tile_dim2=0)
  g1[5] = Kd;                                   // tensor_dim0_stride[31:0]
  g1[6] = 0;                                    // stride hi / dim1_stride lo
  g1[7] = 0;
  v4i z4; z4[0] = 0; z4[1] = 0; z4[2] = 0; z4[3] = 0;
  v8i z8; z8[0] = 0; z8[1] = 0; z8[2] = 0; z8[3] = 0;
  z8[4] = 0; z8[5] = 0; z8[6] = 0; z8[7] = 0;
  __builtin_amdgcn_tensor_load_to_lds(g0, g1, z4, z4, z8, 0);
}
#endif

// ---------------------------------------------------------------------------
// bf16 GEMM, fp32 accumulation: C(MxN) = A(MxK) * BT(NxK)^T, A/BT row-major.
// 256 threads (8 wave32 waves) per 128x128 C tile; each wave owns 32x64 =
// 2x4 v_wmma_f32_16x16x32_bf16 accumulators. A and B tiles are shape-identical
// (128 rows x 32 contiguous K) thanks to the pre-transposed weights.
// TDM path: double-buffered LDS, wave 0 issues tensor_load_to_lds per tile,
// pipelined one tile ahead, synchronized with s_wait_tensorcnt + barriers.
// ---------------------------------------------------------------------------
__global__ __launch_bounds__(256)
void gemm_bf16_kernel(const unsigned short* __restrict__ A,
                      const unsigned short* __restrict__ BT,
                      float* __restrict__ C,
                      int M, int N, int Kd) {
#if USE_TDM
  __shared__ __align__(16) unsigned short As[2][128][40];
  __shared__ __align__(16) unsigned short Bs[2][128][40];
#else
  __shared__ __align__(16) unsigned short As[1][128][40];
  __shared__ __align__(16) unsigned short Bs[1][128][40];
#endif

  const int tid  = threadIdx.x;
  const int lane = tid & 31;
  const int w    = tid >> 5;
  const int wm   = w & 3;          // 0..3 -> 32-row slice
  const int wn   = w >> 2;         // 0..1 -> 64-col slice
  const int h    = lane >> 4;      // lane half
  const int l16  = lane & 15;
  const int rowBase = blockIdx.y * 128;
  const int colBase = blockIdx.x * 128;

  v8f acc[2][4];
#pragma unroll
  for (int sm = 0; sm < 2; ++sm)
#pragma unroll
    for (int sn = 0; sn < 4; ++sn)
#pragma unroll
      for (int r = 0; r < 8; ++r) acc[sm][sn][r] = 0.0f;

  // one K-step of WMMA work from LDS buffer ib
  auto compute = [&](int ib) {
    v16bf afrag[2];
#pragma unroll
    for (int sm = 0; sm < 2; ++sm) {
      const unsigned short* p = &As[ib][wm * 32 + sm * 16 + l16][h * 8];
      v8bf lo = *(const v8bf*)(p);
      v8bf hi = *(const v8bf*)(p + 16);
#pragma unroll
      for (int i = 0; i < 8; ++i) { afrag[sm][i] = lo[i]; afrag[sm][i + 8] = hi[i]; }
    }
#pragma unroll
    for (int sn = 0; sn < 4; ++sn) {
      const unsigned short* p = &Bs[ib][wn * 64 + sn * 16 + l16][h * 8];
      v8bf lo = *(const v8bf*)(p);
      v8bf hi = *(const v8bf*)(p + 16);
      v16bf bfrag;
#pragma unroll
      for (int i = 0; i < 8; ++i) { bfrag[i] = lo[i]; bfrag[i + 8] = hi[i]; }
#pragma unroll
      for (int sm = 0; sm < 2; ++sm) {
        acc[sm][sn] = __builtin_amdgcn_wmma_f32_16x16x32_bf16(
            false, afrag[sm], false, bfrag, (short)0, acc[sm][sn], false, false);
      }
    }
  };

#if USE_TDM
  const unsigned short* Abase = A  + (size_t)rowBase * Kd;
  const unsigned short* Bbase = BT + (size_t)colBase * Kd;
  const int nIter = Kd >> 5;
  if (tid < 32) {   // wave 0 drives the Tensor Data Mover
    tdm_issue_tile((unsigned long long)&As[0][0][0], Abase, Kd);
    tdm_issue_tile((unsigned long long)&Bs[0][0][0], Bbase, Kd);
    if (nIter > 1) {
      tdm_issue_tile((unsigned long long)&As[1][0][0], Abase + 32, Kd);
      tdm_issue_tile((unsigned long long)&Bs[1][0][0], Bbase + 32, Kd);
    }
  }
  for (int it = 0; it < nIter; ++it) {
    const int ib = it & 1;
    if (tid < 32) {
      if (it + 1 < nIter) __builtin_amdgcn_s_wait_tensorcnt(2);  // tile it done
      else                __builtin_amdgcn_s_wait_tensorcnt(0);
    }
    __syncthreads();            // tile it visible to all waves
    compute(ib);
    __syncthreads();            // everyone done reading buffer ib
    if (tid < 32 && it + 2 < nIter) {
      tdm_issue_tile((unsigned long long)&As[ib][0][0],
                     Abase + ((size_t)(it + 2) << 5), Kd);
      tdm_issue_tile((unsigned long long)&Bs[ib][0][0],
                     Bbase + ((size_t)(it + 2) << 5), Kd);
    }
  }
#else
  // synchronous fallback: vectorized staging (A and B tiles identical shape)
  const int arow = tid >> 1;            // 0..127
  const int acol = (tid & 1) * 16;      // 0 / 16
  for (int kk = 0; kk < Kd; kk += 32) {
    __syncthreads();
    {
      const unsigned short* ag = A + (size_t)(rowBase + arow) * Kd + kk + acol;
      *(v8bf*)&As[0][arow][acol]     = *(const v8bf*)(ag);
      *(v8bf*)&As[0][arow][acol + 8] = *(const v8bf*)(ag + 8);
      const unsigned short* bg = BT + (size_t)(colBase + arow) * Kd + kk + acol;
      *(v8bf*)&Bs[0][arow][acol]     = *(const v8bf*)(bg);
      *(v8bf*)&Bs[0][arow][acol + 8] = *(const v8bf*)(bg + 8);
    }
    __syncthreads();
    compute(0);
  }
#endif

  // --- store C: lane l16 = col, vgpr r -> row r + 8*h ---
#pragma unroll
  for (int sm = 0; sm < 2; ++sm) {
#pragma unroll
    for (int sn = 0; sn < 4; ++sn) {
      int col   = colBase + wn * 64 + sn * 16 + l16;
      int mbase = rowBase + wm * 32 + sm * 16 + h * 8;
#pragma unroll
      for (int r = 0; r < 8; ++r)
        C[(size_t)(mbase + r) * N + col] = acc[sm][sn][r];
    }
  }
}

// ---------------------------------------------------------------------------
// scores[n,k] = dot(Q[n,:], Kmat[n*8+k,:]) / 32 ; softmax over k -> attn
// ---------------------------------------------------------------------------
__global__ __launch_bounds__(256)
void scores_softmax_kernel(const float* __restrict__ Q,
                           const float* __restrict__ Km,
                           float* __restrict__ attn) {
  int n = blockIdx.x;
  int w = threadIdx.x >> 5;
  int lane = threadIdx.x & 31;
  const float* q  = Q  + (size_t)n * 1024;
  const float* kr = Km + ((size_t)n * 8 + w) * 1024;
  float s = 0.0f;
#pragma unroll 8
  for (int i = lane; i < 1024; i += 32) s += q[i] * kr[i];
#pragma unroll
  for (int off = 16; off > 0; off >>= 1) s += __shfl_xor(s, off, 32);
  __shared__ float sc[8];
  if (lane == 0) sc[w] = s * 0.03125f;   // 1/sqrt(1024)
  __syncthreads();
  if (threadIdx.x == 0) {
    float m = sc[0];
#pragma unroll
    for (int k = 1; k < 8; ++k) m = fmaxf(m, sc[k]);
    float e[8], sum = 0.0f;
#pragma unroll
    for (int k = 0; k < 8; ++k) { e[k] = __expf(sc[k] - m); sum += e[k]; }
    float inv = 1.0f / sum;
#pragma unroll
    for (int k = 0; k < 8; ++k) attn[(size_t)n * 8 + k] = e[k] * inv;
  }
}

// ---------------------------------------------------------------------------
// Amix[n,d] = sum_k attn[n,k] * Akv[n*8+k, d]   (linearity of Wv)
// ---------------------------------------------------------------------------
__global__ void mix_kernel(const unsigned short* __restrict__ Akv,
                           const float* __restrict__ attn,
                           unsigned short* __restrict__ Amix) {
  int idx = blockIdx.x * 256 + threadIdx.x;   // [0, 2048*4096)
  int d = idx & 4095;
  int n = idx >> 12;
  float s = 0.0f;
#pragma unroll
  for (int k = 0; k < 8; ++k)
    s += attn[n * 8 + k] *
         bf16_to_f32(Akv[(((size_t)n * 8 + k) << 12) + d]);
  Amix[idx] = f32_to_bf16(s);
}

// ---------------------------------------------------------------------------
// Fold patches back: out[b,nf,s1,s2,s3] = G[n, d]
// ---------------------------------------------------------------------------
__global__ void fold_kernel(const float* __restrict__ G,
                            float* __restrict__ out) {
  int idx = blockIdx.x * 256 + threadIdx.x;   // [0, 4*64*32768)
  int s3 = idx & 31, s2 = (idx >> 5) & 31, s1 = (idx >> 10) & 31;
  int nf = (idx >> 15) & 63, b = idx >> 21;
  int p1 = s1 >> 2, e1 = s1 & 3;
  int p2 = s2 >> 2, e2 = s2 & 3;
  int p3 = s3 >> 2, e3 = s3 & 3;
  int n = ((b * 8 + p1) * 8 + p2) * 8 + p3;
  int d = ((nf * 4 + e1) * 4 + e2) * 4 + e3;
  out[idx] = G[((size_t)n << 12) + d];
}

// ---------------------------------------------------------------------------
// Host-side launch
// ---------------------------------------------------------------------------
extern "C" void kernel_launch(void* const* d_in, const int* in_sizes, int n_in,
                              void* d_out, int out_size, void* d_ws, size_t ws_size,
                              hipStream_t stream) {
  const float* xp = (const float*)d_in[0];   // (4,64,32,32,32)
  const float* xr = (const float*)d_in[1];   // (4,8,64,32,32,32)
  const float* Wq = (const float*)d_in[2];   // (4096,1024)
  const float* Wk = (const float*)d_in[3];   // (4096,1024)
  const float* Wv = (const float*)d_in[4];   // (4096,4096)
  float* out = (float*)d_out;                // (4,64,32,32,32)

  char* ws = (char*)d_ws;
  // workspace layout (bytes)
  unsigned short* Apred = (unsigned short*)(ws + 0);                 //  16 MB
  unsigned short* Akv   = (unsigned short*)(ws + 16777216u);         // 128 MB
  unsigned short* WqT   = (unsigned short*)(ws + 150994944u);        //   8 MB (1024x4096)
  unsigned short* WkT   = (unsigned short*)(ws + 159383552u);        //   8 MB (1024x4096)
  unsigned short* WvT   = (unsigned short*)(ws + 167772160u);        //  32 MB (4096x4096)
  unsigned short* Amix  = (unsigned short*)(ws + 201326592u);        //  16 MB
  float*          Qf    = (float*)(ws + 218103808u);                 //   8 MB
  float*          Kmat  = (float*)(ws + 226492416u);                 //  64 MB
  float*          attn  = (float*)(ws + 293601280u);                 //  64 KB
  float*          G     = (float*)(ws + 218103808u);                 // alias dead Q/K, 32 MB

  // 1) transpose + convert weights to bf16 (N-major so GEMM B tiles are
  //    contiguous-K, same shape as A tiles)
  transpose_cvt_kernel<<<dim3(1024 / 32, 4096 / 32), dim3(32, 8), 0, stream>>>(
      Wq, WqT, 4096, 1024);
  transpose_cvt_kernel<<<dim3(1024 / 32, 4096 / 32), dim3(32, 8), 0, stream>>>(
      Wk, WkT, 4096, 1024);
  transpose_cvt_kernel<<<dim3(4096 / 32, 4096 / 32), dim3(32, 8), 0, stream>>>(
      Wv, WvT, 4096, 4096);

  // 2) unfold + convert activations
  pack_pred_kernel<<<32768, 256, 0, stream>>>(xp, Apred);
  pack_kv_kernel<<<262144, 256, 0, stream>>>(xr, Akv);

  // 3) Q = Apred @ Wq  (2048 x 1024, K=4096)
  gemm_bf16_kernel<<<dim3(1024 / 128, 2048 / 128), 256, 0, stream>>>(
      Apred, WqT, Qf, 2048, 1024, 4096);
  // 4) Kmat = Akv @ Wk (16384 x 1024, K=4096)
  gemm_bf16_kernel<<<dim3(1024 / 128, 16384 / 128), 256, 0, stream>>>(
      Akv, WkT, Kmat, 16384, 1024, 4096);

  // 5) scores + softmax -> attn (2048 x 8)
  scores_softmax_kernel<<<2048, 256, 0, stream>>>(Qf, Kmat, attn);

  // 6) attention-weighted mix of raw patches (linearity of Wv)
  mix_kernel<<<32768, 256, 0, stream>>>(Akv, attn, Amix);

  // 7) Out = Amix @ Wv (2048 x 4096, K=4096)
  gemm_bf16_kernel<<<dim3(4096 / 128, 2048 / 128), 256, 0, stream>>>(
      Amix, WvT, G, 2048, 4096, 4096);

  // 8) fold back into volume
  fold_kernel<<<32768, 256, 0, stream>>>(G, out);
}
